// GENConv_68161130988271
// MI455X (gfx1250) — matrix operand order repaired and checked by reading, not compiled, not run on hardware
//
#include <hip/hip_runtime.h>

#define NN 16384
#define DD 128
#define KT 128      // adj columns staged per LDS tile
#define MT 32       // rows per workgroup
#define LDP (KT + 8) // padded LDS row stride (halves) -> 272B, bank-conflict free

typedef __attribute__((ext_vector_type(16))) _Float16 v16h;
typedef __attribute__((ext_vector_type(8)))  _Float16 v8h;
typedef __attribute__((ext_vector_type(8)))  float    v8f;
typedef __attribute__((ext_vector_type(4)))  float    v4f;

union AFrag { v16h v; v8h h[2]; };

static __device__ __forceinline__ v8f wmma_f16(v8h alo, v8h ahi, v8h blo, v8h bhi, v8f c) {
  AFrag a, b;
  a.h[0] = alo; a.h[1] = ahi;
  b.h[0] = blo; b.h[1] = bhi;
  // (neg_a, A, neg_b, B, c_mod, C, reuse_a, reuse_b)
  return __builtin_amdgcn_wmma_f32_16x16x32_f16(false, a.v, false, b.v, (short)0, c, false, false);
}

// xT[d][n] = (f16) x[n][d]   -- B-fragment wants contiguous K (=n) per column d
__global__ __launch_bounds__(256) void genconv_xpose_f16(const float* __restrict__ x,
                                                         _Float16* __restrict__ xT) {
  int idx = blockIdx.x * 256 + threadIdx.x;   // 16384*128 elements
  int n = idx & (NN - 1);
  int d = idx >> 14;
  xT[(long)d * NN + n] = (_Float16)x[(long)n * DD + d];
}

// Wc[n][k] = (f16) W[k][n]   -- column-major f16 copy of W for the MLP B-fragment
__global__ __launch_bounds__(256) void genconv_wprep_f16(const float* __restrict__ W,
                                                         _Float16* __restrict__ Wc) {
  int idx = blockIdx.x * 256 + threadIdx.x;   // 128*128 elements
  int k = idx & (DD - 1);
  int n = idx >> 7;
  Wc[idx] = (_Float16)W[(long)k * DD + n];
}

__global__ __launch_bounds__(256, 2)
void genconv_fused(const float* __restrict__ adj,
                   const _Float16* __restrict__ xT,
                   const _Float16* __restrict__ Wc,
                   const float* __restrict__ bias,
                   float* __restrict__ out) {
  __shared__ _Float16 aTile[MT][LDP];   // exp(adj) tile, f16   (8.5 KB)
  __shared__ _Float16 hbuf[MT][DD + 8]; // normalized h, f16    (8.5 KB)
  __shared__ float    red[MT][8];       // row-sum partials
  __shared__ float    inv[MT];          // 1 / rowsum

  const int tid  = threadIdx.x;
  const int lane = tid & 31;
  const int wave = tid >> 5;            // 0..7, owns 16 output columns
  const int srow = tid >> 3;            // staging row 0..31
  const int scg  = tid & 7;             // staging column group (16 floats each)
  const long r0  = (long)blockIdx.x * MT;

  const int nCol  = (wave << 4) + (lane & 15);  // this lane's N column
  const int aRow  = lane & 15;                  // A-fragment row within tile
  const int c0    = (lane & 16) ? 8 : 0;        // A half-chunk: K[c0..c0+7], K[c0+16..c0+23]
  const int kOff  = (lane & 16) ? 8 : 0;        // B half-chunk: K[kOff..kOff+7], K[kOff+16..kOff+23]
  const int mBase = (lane & 16) ? 8 : 0;        // C/D row base per lane half

  v8f acc0 = {}; v8f acc1 = {};
  float psum = 0.f;

  const float*   adjp = adj + (r0 + srow) * (long)NN + (scg << 4);
  const _Float16* xTp = xT + (long)nCol * NN + kOff;

  for (int k0 = 0; k0 < NN; k0 += KT) {
    // ---- stage exp(adj) tile: 16 f32 per thread, coalesced ----
    v4f a0 = *(const v4f*)(adjp + k0);
    v4f a1 = *(const v4f*)(adjp + k0 + 4);
    v4f a2 = *(const v4f*)(adjp + k0 + 8);
    v4f a3 = *(const v4f*)(adjp + k0 + 12);
    if (k0 + KT < NN) {                    // hint next tile into cache
      __builtin_prefetch(adjp + k0 + KT, 0, 1);
      __builtin_prefetch(adjp + k0 + KT + 64, 0, 1);
    }
    float e[16];
    #pragma unroll
    for (int i = 0; i < 4; ++i) {
      e[i]      = __expf(a0[i]);
      e[4 + i]  = __expf(a1[i]);
      e[8 + i]  = __expf(a2[i]);
      e[12 + i] = __expf(a3[i]);
    }
    v8h lo, hi;
    float s = 0.f;
    #pragma unroll
    for (int i = 0; i < 8; ++i) {
      s += e[i] + e[8 + i];
      lo[i] = (_Float16)e[i];
      hi[i] = (_Float16)e[8 + i];
    }
    psum += s;
    *(v8h*)&aTile[srow][(scg << 4)]     = lo;
    *(v8h*)&aTile[srow][(scg << 4) + 8] = hi;
    __syncthreads();

    // ---- 4 k-steps of 32; two 16-row M tiles share each B fragment ----
    #pragma unroll
    for (int kk = 0; kk < KT; kk += 32) {
      v8h blo  = *(const v8h*)(xTp + k0 + kk);
      v8h bhi  = *(const v8h*)(xTp + k0 + kk + 16);
      v8h a0lo = *(const v8h*)&aTile[aRow][kk + c0];
      v8h a0hi = *(const v8h*)&aTile[aRow][kk + c0 + 16];
      v8h a1lo = *(const v8h*)&aTile[16 + aRow][kk + c0];
      v8h a1hi = *(const v8h*)&aTile[16 + aRow][kk + c0 + 16];
      acc0 = wmma_f16(a0lo, a0hi, blo, bhi, acc0);
      acc1 = wmma_f16(a1lo, a1hi, blo, bhi, acc1);
    }
    __syncthreads();
  }

  // ---- row-sum reduction -> 1/sum ----
  red[srow][scg] = psum;
  __syncthreads();
  if (tid < MT) {
    float s = 0.f;
    #pragma unroll
    for (int j = 0; j < 8; ++j) s += red[tid][j];
    inv[tid] = 1.0f / s;
  }
  __syncthreads();

  // ---- scale accumulators by 1/rowsum, restage h as f16 in LDS ----
  v4f i0 = *(const v4f*)&inv[mBase];
  v4f i1 = *(const v4f*)&inv[mBase + 4];
  v4f i2 = *(const v4f*)&inv[16 + mBase];
  v4f i3 = *(const v4f*)&inv[16 + mBase + 4];
  #pragma unroll
  for (int v = 0; v < 4; ++v) {
    hbuf[mBase + v][nCol]          = (_Float16)(acc0[v]     * i0[v]);
    hbuf[mBase + 4 + v][nCol]      = (_Float16)(acc0[4 + v] * i1[v]);
    hbuf[16 + mBase + v][nCol]     = (_Float16)(acc1[v]     * i2[v]);
    hbuf[16 + mBase + 4 + v][nCol] = (_Float16)(acc1[4 + v] * i3[v]);
  }
  __syncthreads();

  // ---- fused MLP: out[32x128] = h @ W + b ----
  v8f o0 = {}; v8f o1 = {};
  const _Float16* wp = Wc + (long)nCol * DD + kOff;
  #pragma unroll
  for (int kk = 0; kk < DD; kk += 32) {
    v8h blo  = *(const v8h*)(wp + kk);
    v8h bhi  = *(const v8h*)(wp + kk + 16);
    v8h a0lo = *(const v8h*)&hbuf[aRow][kk + c0];
    v8h a0hi = *(const v8h*)&hbuf[aRow][kk + c0 + 16];
    v8h a1lo = *(const v8h*)&hbuf[16 + aRow][kk + c0];
    v8h a1hi = *(const v8h*)&hbuf[16 + aRow][kk + c0 + 16];
    o0 = wmma_f16(a0lo, a0hi, blo, bhi, o0);
    o1 = wmma_f16(a1lo, a1hi, blo, bhi, o1);
  }
  float bb = bias[nCol];
  float* op = out + r0 * DD + nCol;
  #pragma unroll
  for (int v = 0; v < 8; ++v) {
    op[(long)(mBase + v) * DD]      = o0[v] + bb;
    op[(long)(16 + mBase + v) * DD] = o1[v] + bb;
  }
}

extern "C" void kernel_launch(void* const* d_in, const int* in_sizes, int n_in,
                              void* d_out, int out_size, void* d_ws, size_t ws_size,
                              hipStream_t stream) {
  const float* x   = (const float*)d_in[0];   // [N, D_IN]
  const float* adj = (const float*)d_in[1];   // [N, N]
  const float* W   = (const float*)d_in[2];   // [D_IN, D_OUT]
  const float* b   = (const float*)d_in[3];   // [D_OUT]
  float* out = (float*)d_out;

  _Float16* xT = (_Float16*)d_ws;                                   // 4 MB: [D][N] f16
  _Float16* Wc = (_Float16*)((char*)d_ws + (size_t)NN * DD * 2);    // 32 KB: [D_OUT][D_IN] f16

  genconv_xpose_f16<<<(NN * DD) / 256, 256, 0, stream>>>(x, xT);
  genconv_wprep_f16<<<(DD * DD) / 256, 256, 0, stream>>>(W, Wc);
  genconv_fused<<<NN / MT, 256, 0, stream>>>(adj, xT, Wc, b, out);
}
